// TemporalSpatialHamiltonianPredictor_14207751815625
// MI455X (gfx1250) — matrix operand order repaired and checked by reading, not compiled, not run on hardware
//
#include <hip/hip_runtime.h>
#include <hip/hip_bf16.h>
#include <math.h>

#define USE_ASYNC_LDS 1   // gfx1250 GLOBAL_LOAD_ASYNC_TO_LDS_* staging (ASYNCcnt)

// ---------------- types ----------------
typedef __bf16 bf16;
typedef __attribute__((ext_vector_type(16))) __bf16 v16bf;
typedef __attribute__((ext_vector_type(8)))  __bf16 v8bf;
typedef __attribute__((ext_vector_type(8)))  float  v8f;

#define DBT   512        // B*T tokens
#define DIMD  512        // model dim
#define NHD   8
#define HDIM  64
#define FFD   2048
#define EHD   256
#define TTT   32
#define PROJK 65568
#define C1K   160        // 16*9=144 padded to mult of 32
#define C2K   576        // 64*9
#define CBI   64         // images per conv chunk
#define CHR   (CBI*1024) // rows per conv chunk
#define NPIXL (512LL*1024)
#define BK 32

__device__ __forceinline__ float b2f(bf16 x){
  unsigned short h = __builtin_bit_cast(unsigned short, x);
  unsigned u = ((unsigned)h) << 16;
  return __builtin_bit_cast(float, u);
}
__device__ __forceinline__ bf16 f2b(float f){
  unsigned u = __builtin_bit_cast(unsigned, f);
  u += 0x7FFFu + ((u >> 16) & 1u);
  unsigned short h = (unsigned short)(u >> 16);
  return __builtin_bit_cast(bf16, h);
}
__device__ __forceinline__ void async_fence(){
#if USE_ASYNC_LDS
  asm volatile("s_wait_asynccnt 0" ::: "memory");
#endif
}

// Stage a ROWSx32 bf16 tile into LDS. Fast path: per-lane async DMA
// global->LDS (CDNA5 GLOBAL_LOAD_ASYNC_TO_LDS, tracked by ASYNCcnt).
template<int ROWS, int THR>
__device__ __forceinline__ void stage_tile(bf16* __restrict__ dst0,
    const bf16* __restrict__ src, int tid, int ld,
    long long rbase, int rlim, int k0, int K)
{
  constexpr int CH  = ROWS * BK / THR;   // bf16 per thread: 4, 8 or 16
  constexpr int TPR = BK / CH;
  const int r  = tid / TPR;
  const int kk = (tid % TPR) * CH;
  bf16* dst = dst0 + r * BK + kk;
  const long long gr = rbase + r;
  if (gr < rlim && (k0 + kk + CH - 1) < K) {
    const bf16* s = src + gr * (long long)ld + (k0 + kk);
#if USE_ASYNC_LDS
    const unsigned l0 = (unsigned)(unsigned long long)(void*)dst;
    const unsigned long long g0 = (unsigned long long)s;
    if constexpr (CH == 4) {
      asm volatile("global_load_async_to_lds_b64 %0, %1, off" :: "v"(l0), "v"(g0) : "memory");
    } else if constexpr (CH == 8) {
      asm volatile("global_load_async_to_lds_b128 %0, %1, off" :: "v"(l0), "v"(g0) : "memory");
    } else {
      asm volatile("global_load_async_to_lds_b128 %0, %1, off" :: "v"(l0), "v"(g0) : "memory");
      asm volatile("global_load_async_to_lds_b128 %0, %1, off offset:16" :: "v"(l0), "v"(g0) : "memory");
    }
#else
    if constexpr (CH == 4)      *(uint2*)dst = *(const uint2*)s;
    else if constexpr (CH == 8) *(uint4*)dst = *(const uint4*)s;
    else { *(uint4*)dst = *(const uint4*)s; *(uint4*)(dst + 8) = *(const uint4*)(s + 8); }
#endif
  } else {
    for (int j = 0; j < CH; ++j) {
      int kq = k0 + kk + j;
      dst[j] = (gr < rlim && kq < K) ? src[gr * (long long)ld + kq] : f2b(0.f);
    }
  }
}

// =====================================================================
// WMMA GEMM:  C[M,N] = alpha * A[M,K] * B[N,K]^T (+ bias[N]) (+ beta*C)
// bf16 in, fp32 out. Batched with two-level (outer,inner) strides so
// attention heads are addressed in place inside [BT,512] buffers.
// Tile shape templated; every wave owns a WMMA tile in all configs.
// =====================================================================
template<int BMT, int BNT, int WMT, int WNT, int THR>
__global__ __launch_bounds__(THR)
void k_gemm_bf16(const bf16* __restrict__ A, const bf16* __restrict__ B,
                 float* __restrict__ C, const float* __restrict__ bias,
                 int M, int N, int K, int lda, int ldb, int ldc, int nhb,
                 long long sAo, long long sAi, long long sBo, long long sBi,
                 long long sCo, long long sCi, float alpha, float beta)
{
  constexpr int MT = BMT / (16 * WMT);
  constexpr int NT = BNT / (16 * WNT);
  static_assert(WMT * WNT * 32 == THR, "all waves active");

  const int bz = blockIdx.z;
  const int bo = bz / nhb, bi = bz % nhb;
  A += bo * sAo + bi * sAi;
  B += bo * sBo + bi * sBi;
  C += bo * sCo + bi * sCi;

  __shared__ bf16 sA[2][BMT][BK];
  __shared__ bf16 sB[2][BNT][BK];

  const int tid  = threadIdx.x;
  const int lane = tid & 31;
  const int wave = tid >> 5;
  const int wm   = wave % WMT;
  const int wn   = wave / WMT;
  const int m0   = blockIdx.x * BMT;
  const int n0   = blockIdx.y * BNT;

  v8f acc[MT][NT] = {};

  auto loadA = [&](int buf, int k0){
    stage_tile<BMT, THR>(&sA[buf][0][0], A, tid, lda, m0, M, k0, K); };
  auto loadB = [&](int buf, int k0){
    stage_tile<BNT, THR>(&sB[buf][0][0], B, tid, ldb, n0, N, k0, K); };

  const int nk = (K + BK - 1) / BK;
  loadA(0, 0); loadB(0, 0);
  async_fence();
  __syncthreads();

  const int kg = lane >> 4;     // K-group per ISA fragment layout
  const int lr = lane & 15;

  for (int kt = 0; kt < nk; ++kt) {
    const int cur = kt & 1, nxt = cur ^ 1;
    if (kt + 2 < nk) { // L2 prefetch of the tile after next
      __builtin_prefetch(A + (long long)(m0 + (tid & (BMT - 1))) * lda + (kt + 2) * BK, 0, 1);
      __builtin_prefetch(B + (long long)(n0 + (tid & (BNT - 1))) * ldb + (kt + 2) * BK, 0, 1);
    }
    if (kt + 1 < nk) { loadA(nxt, (kt + 1) * BK); loadB(nxt, (kt + 1) * BK); }

    v16bf af[MT];
    v16bf bv[NT];
#pragma unroll
    for (int mt = 0; mt < MT; ++mt) {
      // A 16x32 bf16 frag: lane group kg holds K = kg*8..+7 and kg*8+16..+23
      const v8bf* p0 = (const v8bf*)&sA[cur][wm * 16 * MT + mt * 16 + lr][kg * 8];
      const v8bf* p1 = (const v8bf*)&sA[cur][wm * 16 * MT + mt * 16 + lr][kg * 8 + 16];
      af[mt] = __builtin_shufflevector(*p0, *p1, 0,1,2,3,4,5,6,7,8,9,10,11,12,13,14,15);
    }
#pragma unroll
    for (int nt = 0; nt < NT; ++nt) {
      // B 32x16 frag: lane n=lr, lane group kg holds K = kg*8..+7 and kg*8+16..+23
      const v8bf* q0 = (const v8bf*)&sB[cur][wn * 16 * NT + nt * 16 + lr][kg * 8];
      const v8bf* q1 = (const v8bf*)&sB[cur][wn * 16 * NT + nt * 16 + lr][kg * 8 + 16];
      bv[nt] = __builtin_shufflevector(*q0, *q1, 0,1,2,3,4,5,6,7,8,9,10,11,12,13,14,15);
    }
#pragma unroll
    for (int mt = 0; mt < MT; ++mt)
#pragma unroll
      for (int nt = 0; nt < NT; ++nt)
        acc[mt][nt] = __builtin_amdgcn_wmma_f32_16x16x32_bf16(
            false, af[mt], false, bv[nt], (short)0, acc[mt][nt], false, false);
    async_fence();
    __syncthreads();
  }

  // C layout: VGPR r -> row r (lanes 0-15) / row 8+r (lanes 16-31), col = lane&15
  const int rh = lane >> 4;
#pragma unroll
  for (int mt = 0; mt < MT; ++mt)
#pragma unroll
    for (int nt = 0; nt < NT; ++nt)
#pragma unroll
      for (int r = 0; r < 8; ++r) {
        int row = m0 + wm * 16 * MT + mt * 16 + rh * 8 + r;
        int col = n0 + wn * 16 * NT + nt * 16 + lr;
        if (row < M && col < N) {
          float v = alpha * acc[mt][nt][r];
          if (bias) v += bias[col];
          long long idx = (long long)row * ldc + col;
          if (beta != 0.f) v += beta * C[idx];
          C[idx] = v;
        }
      }
}

// ---------------- elementwise / reshape kernels ----------------
__global__ void k_cvt_bf(const float* __restrict__ s, bf16* __restrict__ d, long long n){
  long long i = blockIdx.x * (long long)blockDim.x + threadIdx.x;
  if (i < n) d[i] = f2b(s[i]);
}
__global__ void k_cvt_pad(const float* __restrict__ s, bf16* __restrict__ d,
                          int R, int Csrc, int Cp, long long n){
  long long i = blockIdx.x * (long long)blockDim.x + threadIdx.x;
  if (i >= n) return;
  int r = (int)(i / Cp), c = (int)(i % Cp);
  d[i] = f2b((r < R && c < Csrc) ? s[(long long)r * Csrc + c] : 0.f);
}
// dst bf16 [Cp][R] = src^T (zero pad rows >= Csrc)
__global__ void k_transpose_cvt(const float* __restrict__ src, bf16* __restrict__ dst,
                                int R, int Csrc, int Cp){
  __shared__ float t[32][33];
  int c0 = blockIdx.x * 32, r0 = blockIdx.y * 32;
  int tx = threadIdx.x, ty = threadIdx.y;
  for (int i = ty; i < 32; i += 8){
    int r = r0 + i, c = c0 + tx;
    t[i][tx] = (r < R && c < Csrc) ? src[(long long)r * Csrc + c] : 0.f;
  }
  __syncthreads();
  for (int i = ty; i < 32; i += 8){
    int c = c0 + i, r = r0 + tx;
    if (c < Cp && r < R) dst[(long long)c * R + r] = f2b(t[tx][i]);
  }
}
// batched small bf16 transpose with two-level src strides
__global__ void k_tr_bf(const bf16* __restrict__ src, bf16* __restrict__ dst,
                        int R, int C, int ldsrc, int nhb,
                        long long sSo, long long sSi, long long dstride){
  int b = blockIdx.y;
  const bf16* s = src + (long long)(b / nhb) * sSo + (long long)(b % nhb) * sSi;
  bf16* d = dst + (long long)b * dstride;
  int i = blockIdx.x * blockDim.x + threadIdx.x;
  if (i >= R * C) return;
  int r = i / C, c = i % C;
  d[(long long)c * R + r] = s[(long long)r * ldsrc + c];
}
__global__ void k_im2col_f32(const float* __restrict__ src, bf16* __restrict__ col,
                             int Cin, int Kc, int btStart, long long n){
  long long i = blockIdx.x * (long long)blockDim.x + threadIdx.x;
  if (i >= n) return;
  int k = (int)(i % Kc); long long row = i / Kc;
  int p = (int)(row & 1023); int bt = btStart + (int)(row >> 10);
  float v = 0.f;
  if (k < Cin * 9){
    int c = k / 9, j = k % 9, ky = j / 3, kx = j % 3;
    int y = p >> 5, x = p & 31, sy = y + ky - 1, sx = x + kx - 1;
    if ((unsigned)sy < 32u && (unsigned)sx < 32u)
      v = src[(((long long)bt * Cin + c) << 10) + (sy << 5) + sx];
  }
  col[i] = f2b(v);
}
__global__ void k_im2col_bf(const bf16* __restrict__ src, bf16* __restrict__ col,
                            int Cin, int Kc, int btStart, long long n){
  long long i = blockIdx.x * (long long)blockDim.x + threadIdx.x;
  if (i >= n) return;
  int k = (int)(i % Kc); long long row = i / Kc;
  int p = (int)(row & 1023); int bt = btStart + (int)(row >> 10);
  bf16 v = f2b(0.f);
  if (k < Cin * 9){
    int c = k / 9, j = k % 9, ky = j / 3, kx = j % 3;
    int y = p >> 5, x = p & 31, sy = y + ky - 1, sx = x + kx - 1;
    if ((unsigned)sy < 32u && (unsigned)sx < 32u)
      v = src[(((long long)bt * Cin + c) << 10) + (sy << 5) + sx];
  }
  col[i] = v;
}
// rows [nbt*1024, 64] fp32 -> leaky -> bf16 (bt, c, p) with token stride
__global__ void k_leaky_perm(const float* __restrict__ g, bf16* __restrict__ dst,
                             long long tokStride, int btStart, long long n){
  long long i = blockIdx.x * (long long)blockDim.x + threadIdx.x;
  if (i >= n) return;
  int c = (int)(i & 63); long long row = i >> 6;
  int p = (int)(row & 1023); int bt = btStart + (int)(row >> 10);
  float v = g[i];
  v = v > 0.f ? v : 0.2f * v;
  dst[(long long)bt * tokStride + ((long long)c << 10) + p] = f2b(v);
}
__global__ void k_fill_aemb(const float* __restrict__ a, bf16* __restrict__ xin, long long n){
  long long i = blockIdx.x * (long long)blockDim.x + threadIdx.x;
  if (i >= n) return;
  int bt = (int)(i >> 5), j = (int)(i & 31);
  xin[(long long)bt * PROJK + 65536 + j] = f2b(a[i]);
}
__global__ void k_add_pos(float* __restrict__ h, const float* __restrict__ pos, long long n){
  long long i = blockIdx.x * (long long)blockDim.x + threadIdx.x;
  if (i >= n) return;
  int bt = (int)(i >> 9), d = (int)(i & 511);
  h[i] += pos[(bt & 31) * DIMD + d];
}
__global__ void k_add(const float* __restrict__ a, const float* __restrict__ b,
                      float* __restrict__ d, long long n){
  long long i = blockIdx.x * (long long)blockDim.x + threadIdx.x;
  if (i < n) d[i] = a[i] + b[i];
}
__global__ __launch_bounds__(256)
void k_ln_fwd(const float* __restrict__ x, const float* __restrict__ g,
              const float* __restrict__ b, bf16* __restrict__ ybf,
              float* __restrict__ xhat, float* __restrict__ rstd, int Dd){
  int row = blockIdx.x; int tid = threadIdx.x;
  const float* xr = x + (long long)row * Dd;
  __shared__ float red[256];
  float s = 0.f;
  for (int j = tid; j < Dd; j += 256) s += xr[j];
  red[tid] = s; __syncthreads();
  for (int st = 128; st > 0; st >>= 1){ if (tid < st) red[tid] += red[tid + st]; __syncthreads(); }
  float mean = red[0] / Dd; __syncthreads();
  float v = 0.f;
  for (int j = tid; j < Dd; j += 256){ float d = xr[j] - mean; v += d * d; }
  red[tid] = v; __syncthreads();
  for (int st = 128; st > 0; st >>= 1){ if (tid < st) red[tid] += red[tid + st]; __syncthreads(); }
  float rs = rsqrtf(red[0] / Dd + 1e-5f);
  if (tid == 0) rstd[row] = rs;
  for (int j = tid; j < Dd; j += 256){
    float xh = (xr[j] - mean) * rs;
    xhat[(long long)row * Dd + j] = xh;
    ybf[(long long)row * Dd + j] = f2b(xh * g[j] + b[j]);
  }
}
__global__ __launch_bounds__(256)
void k_ln_bwd(const float* __restrict__ dy, const float* __restrict__ xhat,
              const float* __restrict__ rstd, const float* __restrict__ g,
              float* __restrict__ dx_io, int Dd){
  int row = blockIdx.x; int tid = threadIdx.x;
  __shared__ float r1[256], r2[256];
  float s1 = 0.f, s2 = 0.f;
  for (int j = tid; j < Dd; j += 256){
    float a = dy[(long long)row * Dd + j] * g[j];
    float xh = xhat[(long long)row * Dd + j];
    s1 += a; s2 += a * xh;
  }
  r1[tid] = s1; r2[tid] = s2; __syncthreads();
  for (int st = 128; st > 0; st >>= 1){
    if (tid < st){ r1[tid] += r1[tid + st]; r2[tid] += r2[tid + st]; }
    __syncthreads();
  }
  float m1 = r1[0] / Dd, m2 = r2[0] / Dd, rs = rstd[row];
  for (int j = tid; j < Dd; j += 256){
    float a = dy[(long long)row * Dd + j] * g[j];
    float xh = xhat[(long long)row * Dd + j];
    dx_io[(long long)row * Dd + j] += rs * (a - m1 - xh * m2);
  }
}
__global__ void k_softmax_fwd(float* __restrict__ sc, bf16* __restrict__ pbf){
  int rowg = blockIdx.x;                 // (b*8+h)*32 + i
  int i = rowg & 31, j = threadIdx.x;
  float* r = sc + (long long)rowg * TTT;
  float v = (j <= i) ? r[j] : -3.0e38f;
  float m = v;
  for (int o = 16; o > 0; o >>= 1) m = fmaxf(m, __shfl_xor(m, o, 32));
  float e = (j <= i) ? __expf(v - m) : 0.f;
  float s = e;
  for (int o = 16; o > 0; o >>= 1) s += __shfl_xor(s, o, 32);
  float p = e / s;
  r[j] = p;
  pbf[(long long)rowg * TTT + j] = f2b(p);
}
__global__ void k_softmax_bwd(const float* __restrict__ dP, const float* __restrict__ P,
                              bf16* __restrict__ dSbf, float scale){
  int rowg = blockIdx.x; int j = threadIdx.x;
  float p = P[(long long)rowg * TTT + j], dp = dP[(long long)rowg * TTT + j];
  float s = dp * p;
  for (int o = 16; o > 0; o >>= 1) s += __shfl_xor(s, o, 32);
  dSbf[(long long)rowg * TTT + j] = f2b(p * (dp - s) * scale);
}
__global__ void k_gelu_fwd(const float* __restrict__ x, bf16* __restrict__ o, long long n){
  long long i = blockIdx.x * (long long)blockDim.x + threadIdx.x;
  if (i >= n) return;
  float v = x[i];
  o[i] = f2b(0.5f * v * (1.f + erff(v * 0.70710678f)));
}
__global__ void k_gelu_bwd(const float* __restrict__ dg, const float* __restrict__ x,
                           bf16* __restrict__ o, long long n){
  long long i = blockIdx.x * (long long)blockDim.x + threadIdx.x;
  if (i >= n) return;
  float v = x[i];
  float d = 0.5f * (1.f + erff(v * 0.70710678f)) + v * __expf(-0.5f * v * v) * 0.39894228f;
  o[i] = f2b(dg[i] * d);
}
__global__ void k_sigscale(const float* __restrict__ s, const float* __restrict__ w2,
                           bf16* __restrict__ o, long long n){
  long long i = blockIdx.x * (long long)blockDim.x + threadIdx.x;
  if (i >= n) return;
  o[i] = f2b(w2[i & (EHD - 1)] / (1.f + __expf(-s[i])));
}
// (bt,c,p) grad * leaky-mask(act) -> row-major [bt*1024+p, 64] bf16
__global__ void k_leaky_bwd(const float* __restrict__ dsrc, long long tds,
                            const bf16* __restrict__ act, long long tas,
                            bf16* __restrict__ rows, long long n){
  long long i = blockIdx.x * (long long)blockDim.x + threadIdx.x;
  if (i >= n) return;
  int p = (int)(i & 1023); long long r = i >> 10;
  int c = (int)(r & 63); int bt = (int)(r >> 6);
  long long src = (long long)bt * tds + ((long long)c << 10) + p;
  float d = dsrc[src];
  float a = b2f(act[(long long)bt * tas + ((long long)c << 10) + p]);
  if (a <= 0.f) d *= 0.2f;
  rows[(((long long)bt << 10) + p) * 64 + c] = f2b(d);
}
__global__ void k_col2im(const float* __restrict__ col, float* __restrict__ dx,
                         int Cin, int Kc, int btStart, long long n){
  long long i = blockIdx.x * (long long)blockDim.x + threadIdx.x;
  if (i >= n) return;
  int p = (int)(i & 1023); long long r = i >> 10;
  int c = (int)(r % Cin); int btl = (int)(r / Cin);
  int y = p >> 5, x = p & 31;
  float s = 0.f;
  for (int ky = 0; ky < 3; ++ky){
    int py = y - ky + 1; if ((unsigned)py >= 32u) continue;
    for (int kx = 0; kx < 3; ++kx){
      int px = x - kx + 1; if ((unsigned)px >= 32u) continue;
      s += col[(((long long)btl << 10) + (py << 5) + px) * Kc + c * 9 + ky * 3 + kx];
    }
  }
  dx[(((long long)(btStart + btl) * Cin + c) << 10) + p] = s;
}
__global__ void k_gu_final(const float* __restrict__ dz, const float* __restrict__ as,
                           const float* __restrict__ Gw, const float* __restrict__ Gb,
                           const float* __restrict__ damp, float* __restrict__ out, long long n){
  long long i = blockIdx.x * (long long)blockDim.x + threadIdx.x;
  if (i >= n) return;
  int p = (int)(i & 1023); long long r = i >> 10;
  int o = (int)(r & 7); int bt = (int)(r >> 3);
  float gu = Gb[o];
  for (int c = 0; c < 32; ++c)
    gu += as[(((long long)bt * 32 + c) << 10) + p] * Gw[o * 32 + c];
  float dq = dz[(((long long)bt * 16 + o) << 10) + p];
  float dp = dz[(((long long)bt * 16 + 8 + o) << 10) + p];
  out[(((long long)bt * 16 + o) << 10) + p]     = dp;                         // dq_dt
  out[(((long long)bt * 16 + 8 + o) << 10) + p] = -dq - damp[0] * dp + gu;    // dp_dt
}

// ---------------- host side ----------------
static inline void gemm(hipStream_t st, const bf16* A, const bf16* B, float* C,
                        const float* bias, int M, int N, int K,
                        int lda, int ldb, int ldc,
                        int batch = 1, int nhb = 1,
                        long long sAo = 0, long long sAi = 0,
                        long long sBo = 0, long long sBi = 0,
                        long long sCo = 0, long long sCi = 0,
                        float alpha = 1.f, float beta = 0.f){
  if (M <= 32 && N <= 32) {
    dim3 g(1, 1, batch);
    k_gemm_bf16<32, 32, 2, 2, 128><<<g, dim3(128), 0, st>>>(A, B, C, bias, M, N, K,
        lda, ldb, ldc, nhb, sAo, sAi, sBo, sBi, sCo, sCi, alpha, beta);
  } else if (M <= 32) {
    dim3 g(1, (N + 63) / 64, batch);
    k_gemm_bf16<32, 64, 2, 4, 256><<<g, dim3(256), 0, st>>>(A, B, C, bias, M, N, K,
        lda, ldb, ldc, nhb, sAo, sAi, sBo, sBi, sCo, sCi, alpha, beta);
  } else if (N <= 64) {
    dim3 g((M + 127) / 128, 1, batch);
    k_gemm_bf16<128, 64, 8, 1, 256><<<g, dim3(256), 0, st>>>(A, B, C, bias, M, N, K,
        lda, ldb, ldc, nhb, sAo, sAi, sBo, sBi, sCo, sCi, alpha, beta);
  } else {
    dim3 g((M + 127) / 128, (N + 127) / 128, batch);
    k_gemm_bf16<128, 128, 4, 2, 256><<<g, dim3(256), 0, st>>>(A, B, C, bias, M, N, K,
        lda, ldb, ldc, nhb, sAo, sAi, sBo, sBi, sCo, sCi, alpha, beta);
  }
}
#define EL(kern, n, ...) do { long long _n = (long long)(n); \
  unsigned _g = (unsigned)((_n + 255) / 256); \
  kern<<<dim3(_g), dim3(256), 0, stream>>>(__VA_ARGS__); } while (0)

extern "C" void kernel_launch(void* const* d_in, const int* in_sizes, int n_in,
                              void* d_out, int out_size, void* d_ws, size_t ws_size,
                              hipStream_t stream)
{
  (void)in_sizes; (void)n_in; (void)out_size; (void)ws_size;
  // Input leaf order: setup_inputs() dict insertion order, params flattened
  // recursively in written order (layers list in order, each layer dict in
  // written order).
  const float* z       = (const float*)d_in[0];
  const float* aspat   = (const float*)d_in[1];
  const float* aemb    = (const float*)d_in[2];
  const float* damp    = (const float*)d_in[3];
  const float* conv1_w = (const float*)d_in[4];
  const float* conv1_b = (const float*)d_in[5];
  const float* conv2_w = (const float*)d_in[6];
  const float* conv2_b = (const float*)d_in[7];
  const float* proj_w  = (const float*)d_in[8];
  const float* proj_b  = (const float*)d_in[9];
  const float* pos     = (const float*)d_in[10];
  struct Lp { const float *ln1_g,*ln1_b,*q_w,*q_b,*k_w,*k_b,*v_w,*v_b,
                          *o_w,*o_b,*ln2_g,*ln2_b,*ff1_w,*ff1_b,*ff2_w,*ff2_b; } L[2];
  int ix = 11;
  for (int l = 0; l < 2; ++l){
    const float** f = (const float**)&L[l];
    for (int j = 0; j < 16; ++j) f[j] = (const float*)d_in[ix++];
  }
  const float* T_w1 = (const float*)d_in[ix++];
  const float* T_b1 = (const float*)d_in[ix++];
  const float* T_w2 = (const float*)d_in[ix++]; ix++; // T_b2 unused (sum-grad)
  const float* V_w1 = (const float*)d_in[ix++];
  const float* V_b1 = (const float*)d_in[ix++];
  const float* V_w2 = (const float*)d_in[ix++]; ix++; // V_b2 unused
  const float* G_w  = (const float*)d_in[ix++];
  const float* G_b  = (const float*)d_in[ix++];

  char* wsb = (char*)d_ws; size_t off = 0;
  auto alloc = [&](size_t bytes)->void*{ void* p = wsb + off;
                                         off = (off + bytes + 255) & ~(size_t)255; return p; };
  auto abf = [&](long long n){ return (bf16*)alloc((size_t)n * 2); };
  auto af  = [&](long long n){ return (float*)alloc((size_t)n * 4); };

  // ---- weights -> bf16 (+ transposed copies for the backward GEMMs) ----
  bf16* c1w  = abf(64 * C1K);
  EL(k_cvt_pad, 64 * C1K, conv1_w, c1w, 64, 144, C1K, (long long)64 * C1K);
  bf16* c1wT = abf(C1K * 64);
  { dim3 g((C1K + 31) / 32, 2); k_transpose_cvt<<<g, dim3(32, 8), 0, stream>>>(conv1_w, c1wT, 64, 144, C1K); }
  bf16* c2w  = abf(64 * C2K);
  EL(k_cvt_bf, 64 * C2K, conv2_w, c2w, (long long)64 * C2K);
  bf16* c2wT = abf(C2K * 64);
  { dim3 g((C2K + 31) / 32, 2); k_transpose_cvt<<<g, dim3(32, 8), 0, stream>>>(conv2_w, c2wT, 64, C2K, C2K); }
  bf16* pw   = abf((long long)DIMD * PROJK);
  EL(k_cvt_bf, (long long)DIMD * PROJK, proj_w, pw, (long long)DIMD * PROJK);
  bf16* pwT  = abf((long long)PROJK * DIMD);
  { dim3 g((PROJK + 31) / 32, (DIMD + 31) / 32);
    k_transpose_cvt<<<g, dim3(32, 8), 0, stream>>>(proj_w, pwT, DIMD, PROJK, PROJK); }

  bf16 *qw[2],*kw[2],*vw[2],*ow[2],*qwT[2],*kwT[2],*vwT[2],*owT[2];
  bf16 *f1w[2],*f1wT[2],*f2w[2],*f2wT[2];
  for (int l = 0; l < 2; ++l){
    const float* W4[4]  = { L[l].q_w, L[l].k_w, L[l].v_w, L[l].o_w };
    bf16** P4[4]  = { &qw[l], &kw[l], &vw[l], &ow[l] };
    bf16** PT4[4] = { &qwT[l], &kwT[l], &vwT[l], &owT[l] };
    for (int j = 0; j < 4; ++j){
      *P4[j] = abf((long long)DIMD * DIMD);
      EL(k_cvt_bf, (long long)DIMD * DIMD, W4[j], *P4[j], (long long)DIMD * DIMD);
      *PT4[j] = abf((long long)DIMD * DIMD);
      dim3 g(16, 16); k_transpose_cvt<<<g, dim3(32, 8), 0, stream>>>(W4[j], *PT4[j], DIMD, DIMD, DIMD);
    }
    f1w[l]  = abf((long long)FFD * DIMD);
    EL(k_cvt_bf, (long long)FFD * DIMD, L[l].ff1_w, f1w[l], (long long)FFD * DIMD);
    f1wT[l] = abf((long long)DIMD * FFD);
    { dim3 g(16, 64); k_transpose_cvt<<<g, dim3(32, 8), 0, stream>>>(L[l].ff1_w, f1wT[l], FFD, DIMD, DIMD); }
    f2w[l]  = abf((long long)DIMD * FFD);
    EL(k_cvt_bf, (long long)DIMD * FFD, L[l].ff2_w, f2w[l], (long long)DIMD * FFD);
    f2wT[l] = abf((long long)FFD * DIMD);
    { dim3 g(64, 16); k_transpose_cvt<<<g, dim3(32, 8), 0, stream>>>(L[l].ff2_w, f2wT[l], DIMD, FFD, FFD); }
  }
  bf16* tw1  = abf(EHD * EHD); EL(k_cvt_bf, EHD * EHD, T_w1, tw1, (long long)EHD * EHD);
  bf16* tw1T = abf(EHD * EHD);
  { dim3 g(8, 8); k_transpose_cvt<<<g, dim3(32, 8), 0, stream>>>(T_w1, tw1T, EHD, EHD, EHD); }
  bf16* vw1  = abf(EHD * EHD); EL(k_cvt_bf, EHD * EHD, V_w1, vw1, (long long)EHD * EHD);
  bf16* vw1T = abf(EHD * EHD);
  { dim3 g(8, 8); k_transpose_cvt<<<g, dim3(32, 8), 0, stream>>>(V_w1, vw1T, EHD, EHD, EHD); }

  // ---- activations / scratch ----
  bf16*  h1act  = abf(NPIXL * 64);
  bf16*  xin    = abf((long long)DBT * PROJK);
  bf16*  colbuf = abf((long long)CHR * C2K);
  float* gbuf   = af((long long)CHR * C2K);
  float* x0 = af(DBT * DIMD), *xA = af(DBT * DIMD), *x2 = af(DBT * DIMD);
  float* xm[2] = { af(DBT * DIMD), af(DBT * DIMD) };
  float *xhat1[2], *rstd1[2], *xhat2[2], *rstd2[2], *probs[2], *sff1[2];
  bf16  *a1bf[2], *a2bf[2], *qbf[2], *kbf[2], *vbf[2], *probsbf[2], *gelbf[2];
  for (int l = 0; l < 2; ++l){
    xhat1[l] = af(DBT * DIMD); rstd1[l] = af(DBT);
    xhat2[l] = af(DBT * DIMD); rstd2[l] = af(DBT);
    a1bf[l] = abf(DBT * DIMD); a2bf[l] = abf(DBT * DIMD);
    qbf[l] = abf(DBT * DIMD); kbf[l] = abf(DBT * DIMD); vbf[l] = abf(DBT * DIMD);
    probs[l] = af(131072); probsbf[l] = abf(131072);
    sff1[l] = af((long long)DBT * FFD); gelbf[l] = abf((long long)DBT * FFD);
  }
  float* tF  = af((long long)DBT * FFD);
  float* tF2 = af(DBT * DIMD);
  float* attnf = af(DBT * DIMD);  bf16* attnbf = abf(DBT * DIMD);
  bf16* vT = abf(262144); bf16* kT = abf(262144); bf16* qT = abf(262144); bf16* datT = abf(262144);
  bf16* pT = abf(131072); bf16* dSbf = abf(131072); bf16* dSTbf = abf(131072);
  float* dPf = af(131072);
  float* dx = af(DBT * DIMD);     bf16* dxbf = abf(DBT * DIMD);
  float* dattn = af(DBT * DIMD);  bf16* dattnbf = abf(DBT * DIMD);
  bf16* gelbwd = abf((long long)DBT * FFD);
  float* dqF = af(DBT * DIMD), *dkF = af(DBT * DIMD), *dvF = af(DBT * DIMD);
  bf16* dqbf = abf(DBT * DIMD), *dkbf = abf(DBT * DIMD), *dvbf = abf(DBT * DIMD);
  float* dln1 = af(DBT * DIMD);
  float* sT = af(DBT * EHD), *sV = af(DBT * EHD);
  bf16* gTbf = abf(DBT * EHD), *gVbf = abf(DBT * EHD);
  bf16* featbf = abf(DBT * DIMD);
  float* dxin = af((long long)DBT * PROJK);
  bf16* dh2rows = abf(NPIXL * 64);
  float* dh1 = af(NPIXL * 64);
  bf16* dh1rows = abf(NPIXL * 64);
  float* dzz = af(512LL * 16 * 1024);

  const long long hTD = (long long)TTT * DIMD;   // per-batch stride in [BT,512]

  // ================= forward: convs (chunked implicit GEMM) =================
  for (int cs = 0; cs < 512; cs += CBI){
    EL(k_im2col_f32, (long long)CHR * C1K, z, colbuf, 16, C1K, cs, (long long)CHR * C1K);
    gemm(stream, colbuf, c1w, gbuf, conv1_b, CHR, 64, C1K, C1K, C1K, 64);
    EL(k_leaky_perm, (long long)CHR * 64, gbuf, h1act, 65536LL, cs, (long long)CHR * 64);
  }
  for (int cs = 0; cs < 512; cs += CBI){
    EL(k_im2col_bf, (long long)CHR * C2K, h1act, colbuf, 64, C2K, cs, (long long)CHR * C2K);
    gemm(stream, colbuf, c2w, gbuf, conv2_b, CHR, 64, C2K, C2K, C2K, 64);
    EL(k_leaky_perm, (long long)CHR * 64, gbuf, xin, (long long)PROJK, cs, (long long)CHR * 64);
  }
  EL(k_fill_aemb, 512 * 32, aemb, xin, 512LL * 32);

  // ================= forward: projection + transformer =================
  gemm(stream, xin, pw, x0, proj_b, DBT, DIMD, PROJK, PROJK, PROJK, DIMD);
  EL(k_add_pos, DBT * DIMD, x0, pos, (long long)DBT * DIMD);

  float* xi = x0;
  for (int l = 0; l < 2; ++l){
    float* xmid = xm[l];
    float* xo = (l == 0) ? xA : x2;
    k_ln_fwd<<<DBT, 256, 0, stream>>>(xi, L[l].ln1_g, L[l].ln1_b, a1bf[l], xhat1[l], rstd1[l], DIMD);
    gemm(stream, a1bf[l], qw[l], tF2, L[l].q_b, DBT, DIMD, DIMD, DIMD, DIMD, DIMD);
    EL(k_cvt_bf, DBT * DIMD, tF2, qbf[l], (long long)DBT * DIMD);
    gemm(stream, a1bf[l], kw[l], tF2, L[l].k_b, DBT, DIMD, DIMD, DIMD, DIMD, DIMD);
    EL(k_cvt_bf, DBT * DIMD, tF2, kbf[l], (long long)DBT * DIMD);
    gemm(stream, a1bf[l], vw[l], tF2, L[l].v_b, DBT, DIMD, DIMD, DIMD, DIMD, DIMD);
    EL(k_cvt_bf, DBT * DIMD, tF2, vbf[l], (long long)DBT * DIMD);
    // scores = (Q K^T)/8  (batched per (b,h); heads live inside [BT,512])
    gemm(stream, qbf[l], kbf[l], probs[l], nullptr, TTT, TTT, HDIM, DIMD, DIMD, TTT,
         128, NHD, hTD, 64, hTD, 64, 8192, 1024, 0.125f, 0.f);
    k_softmax_fwd<<<4096, 32, 0, stream>>>(probs[l], probsbf[l]);
    { dim3 g(8, 128); k_tr_bf<<<g, 256, 0, stream>>>(vbf[l], vT, TTT, HDIM, DIMD, NHD, hTD, 64, 2048); }
    gemm(stream, probsbf[l], vT, attnf, nullptr, TTT, HDIM, TTT, TTT, TTT, DIMD,
         128, NHD, 8192, 1024, 16384, 2048, hTD, 64, 1.f, 0.f);
    EL(k_cvt_bf, DBT * DIMD, attnf, attnbf, (long long)DBT * DIMD);
    gemm(stream, attnbf, ow[l], tF2, L[l].o_b, DBT, DIMD, DIMD, DIMD, DIMD, DIMD);
    EL(k_add, DBT * DIMD, xi, tF2, xmid, (long long)DBT * DIMD);
    k_ln_fwd<<<DBT, 256, 0, stream>>>(xmid, L[l].ln2_g, L[l].ln2_b, a2bf[l], xhat2[l], rstd2[l], DIMD);
    gemm(stream, a2bf[l], f1w[l], sff1[l], L[l].ff1_b, DBT, FFD, DIMD, DIMD, DIMD, FFD);
    EL(k_gelu_fwd, (long long)DBT * FFD, sff1[l], gelbf[l], (long long)DBT * FFD);
    gemm(stream, gelbf[l], f2w[l], tF2, L[l].ff2_b, DBT, DIMD, FFD, FFD, FFD, DIMD);
    EL(k_add, DBT * DIMD, xmid, tF2, xo, (long long)DBT * DIMD);
    xi = xo;
  }

  // ================= head (only pre-softplus acts needed) + its backward ====
  EL(k_cvt_bf, DBT * DIMD, x2, featbf, (long long)DBT * DIMD);
  gemm(stream, featbf + EHD, tw1, sT, T_b1, DBT, EHD, EHD, DIMD, EHD, EHD); // pf branch
  gemm(stream, featbf,       vw1, sV, V_b1, DBT, EHD, EHD, DIMD, EHD, EHD); // qf branch
  EL(k_sigscale, DBT * EHD, sT, T_w2, gTbf, (long long)DBT * EHD);
  EL(k_sigscale, DBT * EHD, sV, V_w2, gVbf, (long long)DBT * EHD);
  gemm(stream, gTbf, tw1T, dx + EHD, nullptr, DBT, EHD, EHD, EHD, EHD, DIMD); // dpf
  gemm(stream, gVbf, vw1T, dx,       nullptr, DBT, EHD, EHD, EHD, EHD, DIMD); // dqf

  // ================= transformer backward =================
  for (int l = 1; l >= 0; --l){
    EL(k_cvt_bf, DBT * DIMD, dx, dxbf, (long long)DBT * DIMD);
    gemm(stream, dxbf, f2wT[l], tF, nullptr, DBT, FFD, DIMD, DIMD, DIMD, FFD);
    EL(k_gelu_bwd, (long long)DBT * FFD, tF, sff1[l], gelbwd, (long long)DBT * FFD);
    gemm(stream, gelbwd, f1wT[l], tF2, nullptr, DBT, DIMD, FFD, FFD, FFD, DIMD);
    k_ln_bwd<<<DBT, 256, 0, stream>>>(tF2, xhat2[l], rstd2[l], L[l].ln2_g, dx, DIMD);
    // dx = grad at x_mid
    EL(k_cvt_bf, DBT * DIMD, dx, dxbf, (long long)DBT * DIMD);
    gemm(stream, dxbf, owT[l], dattn, nullptr, DBT, DIMD, DIMD, DIMD, DIMD, DIMD);
    EL(k_cvt_bf, DBT * DIMD, dattn, dattnbf, (long long)DBT * DIMD);
    { dim3 g(8, 128);
      k_tr_bf<<<g, 256, 0, stream>>>(dattnbf, datT, TTT, HDIM, DIMD, NHD, hTD, 64, 2048);
      k_tr_bf<<<g, 256, 0, stream>>>(kbf[l], kT, TTT, HDIM, DIMD, NHD, hTD, 64, 2048);
      k_tr_bf<<<g, 256, 0, stream>>>(qbf[l], qT, TTT, HDIM, DIMD, NHD, hTD, 64, 2048); }
    { dim3 g(4, 128);
      k_tr_bf<<<g, 256, 0, stream>>>(probsbf[l], pT, TTT, TTT, TTT, NHD, 8192, 1024, 1024); }
    gemm(stream, dattnbf, vbf[l], dPf, nullptr, TTT, TTT, HDIM, DIMD, DIMD, TTT,
         128, NHD, hTD, 64, hTD, 64, 8192, 1024, 1.f, 0.f);
    k_softmax_bwd<<<4096, 32, 0, stream>>>(dPf, probs[l], dSbf, 0.125f);
    { dim3 g(4, 128);
      k_tr_bf<<<g, 256, 0, stream>>>(dSbf, dSTbf, TTT, TTT, TTT, NHD, 8192, 1024, 1024); }
    gemm(stream, dSbf,  kT,   dqF, nullptr, TTT, HDIM, TTT, TTT, TTT, DIMD,
         128, NHD, 8192, 1024, 16384, 2048, hTD, 64, 1.f, 0.f);
    gemm(stream, dSTbf, qT,   dkF, nullptr, TTT, HDIM, TTT, TTT, TTT, DIMD,
         128, NHD, 8192, 1024, 16384, 2048, hTD, 64, 1.f, 0.f);
    gemm(stream, pT,    datT, dvF, nullptr, TTT, HDIM, TTT, TTT, TTT, DIMD,
         128, NHD, 8192, 1024, 16384, 2048, hTD, 64, 1.f, 0.f);
    EL(k_cvt_bf, DBT * DIMD, dqF, dqbf, (long long)DBT * DIMD);
    EL(k_cvt_bf, DBT * DIMD, dkF, dkbf, (long long)DBT * DIMD);
    EL(k_cvt_bf, DBT * DIMD, dvF, dvbf, (long long)DBT * DIMD);
    gemm(stream, dqbf, qwT[l], dln1, nullptr, DBT, DIMD, DIMD, DIMD, DIMD, DIMD);
    gemm(stream, dkbf, kwT[l], dln1, nullptr, DBT, DIMD, DIMD, DIMD, DIMD, DIMD,
         1, 1, 0, 0, 0, 0, 0, 0, 1.f, 1.f);
    gemm(stream, dvbf, vwT[l], dln1, nullptr, DBT, DIMD, DIMD, DIMD, DIMD, DIMD,
         1, 1, 0, 0, 0, 0, 0, 0, 1.f, 1.f);
    k_ln_bwd<<<DBT, 256, 0, stream>>>(dln1, xhat1[l], rstd1[l], L[l].ln1_g, dx, DIMD);
    // dx = grad at layer input
  }

  // ================= projection + conv backward =================
  EL(k_cvt_bf, DBT * DIMD, dx, dxbf, (long long)DBT * DIMD);
  gemm(stream, dxbf, pwT, dxin, nullptr, DBT, PROJK, DIMD, DIMD, DIMD, PROJK);
  EL(k_leaky_bwd, NPIXL * 64, dxin, (long long)PROJK, xin, (long long)PROJK,
     dh2rows, NPIXL * 64);
  for (int cs = 0; cs < 512; cs += CBI){
    gemm(stream, dh2rows + (long long)cs * 1024 * 64, c2wT, gbuf, nullptr,
         CHR, C2K, 64, 64, 64, C2K);
    EL(k_col2im, (long long)CBI * 64 * 1024, gbuf, dh1, 64, C2K, cs, (long long)CBI * 64 * 1024);
  }
  EL(k_leaky_bwd, NPIXL * 64, dh1, 65536LL, h1act, 65536LL, dh1rows, NPIXL * 64);
  for (int cs = 0; cs < 512; cs += CBI){
    gemm(stream, dh1rows + (long long)cs * 1024 * 64, c1wT, gbuf, nullptr,
         CHR, C1K, 64, 64, 64, C1K);
    EL(k_col2im, (long long)CBI * 16 * 1024, gbuf, dzz, 16, C1K, cs, (long long)CBI * 16 * 1024);
  }

  // ================= Hamiltonian dynamics output =================
  EL(k_gu_final, 512LL * 8 * 1024, dzz, aspat, G_w, G_b, damp, (float*)d_out,
     512LL * 8 * 1024);
}
